// RNN_MultiRegional_56040733278456
// MI455X (gfx1250) — compile-verified, matrix operand order checked by reading, not run on hardware
//
#include <hip/hip_runtime.h>
#include <hip/hip_bf16.h>
#include <math.h>

// ---------------------------------------------------------------------------
// Multi-regional RNN scan for MI455X (gfx1250, wave32, WMMA).
//
//   B=128, T=1000, I=64, H=1024, HALF=512.
//   Grid = 64 WGs = 8 clusters (16-row batch tile) x 8 WGs (128-col W slice).
//   Each WG: 8 waves; wave w owns a 16x16 output tile, accumulating over
//   K=1024 (+64 drive K for striatum columns) with v_wmma_f32_16x16x32_f16.
//   W slice resident in LDS (272 KB / 320 KB) for all 1000 steps; per step
//   only the 16x1024 f16 h tile moves through L2; cluster barrier (-3)
//   separates steps.
// ---------------------------------------------------------------------------

typedef __attribute__((ext_vector_type(16))) _Float16 v16h;
typedef __attribute__((ext_vector_type(8)))  float    v8f;

#define BB    128
#define TT    1000
#define II    64
#define HH    1024
#define HHALF 512
#define NEXC  359           // HALF - N_INH

#define NKT_H   32          // K-tiles covering H=1024 (K=32 each)
#define NKT_ALL 34          // + 2 K-tiles covering the I=64 input drive
#define TILE_E  512         // f16 elements per 16x32 packed fragment tile

// d_out float offsets:  out | hn_last | rnn_out | x_last | x_out
#define OUT_OFF    0LL
#define HN_OFF     128000LL
#define RNN_OFF    259072LL
#define XLAST_OFF  131331072LL
#define XOUT_OFF   131462144LL

// workspace layout (bytes)
#define WPACK_ELEMS (64LL * NKT_ALL * TILE_E)        // 64 n-tiles
#define WPACK_BYTES (WPACK_ELEMS * 2)                // 2,228,224
#define HBUF_BYTES  (2LL * BB * HH * 2)              //   524,288
#define INP16_BYTES ((long long)BB * TT * II * 2)    // 16,384,000

#define WLDS_ELEMS (8 * NKT_ALL * TILE_E)            // 139,264 f16
#define ALDS_ELEMS (NKT_ALL * TILE_E)                //  17,408 f16
#define LDS_BYTES  ((WLDS_ELEMS + ALDS_ELEMS) * 2)   // 313,344 B <= 320 KB

// fragment-element -> (lane-row, k) mapping for the 16-bit 16x32 A layout.
// lane L: row = L%16, half = L/16; the lane's 16 halves are two runs of 8
// consecutive K: [half*8, +8) and [16+half*8, +8).
__device__ __forceinline__ int frag_klocal(int half, int tt) {
    return (tt < 8) ? (half * 8 + tt) : (16 + half * 8 + (tt - 8));
}

// ---------------------------------------------------------------------------
// Kernel 1: build packed f16 W (B-fragment order): Wpack[(nt*34+kt)*512 + e]
// ---------------------------------------------------------------------------
__global__ void pack_w_kernel(const float* __restrict__ str2str_fixed,
                              const float* __restrict__ m12str_w,
                              const float* __restrict__ str2m1_w,
                              const float* __restrict__ m12m1_w,
                              const float* __restrict__ inp_weight,
                              _Float16* __restrict__ Wpack) {
    long long idx = (long long)blockIdx.x * blockDim.x + threadIdx.x;
    if (idx >= WPACK_ELEMS) return;
    int e  = (int)(idx & (TILE_E - 1));
    int kt = (int)((idx >> 9) % NKT_ALL);
    int nt = (int)(idx / (TILE_E * NKT_ALL));
    int lane = e >> 4, tt = e & 15;
    int nl = lane & 15, half = lane >> 4;
    int klocal = frag_klocal(half, tt);
    int n = nt * 16 + nl;                       // output unit (row of W_rec)
    float val;
    if (kt < NKT_H) {
        int k = kt * 32 + klocal;               // input unit (col of W_rec)
        if (n < HHALF) {
            if (k < HHALF) {
                val = -str2str_fixed[n * HHALF + k];            // fixed @ (-I)
            } else {
                int c = k - HHALF;
                val = (c < NEXC) ? fmaxf(m12str_w[n * HHALF + c], 0.f) : 0.f;
            }
        } else {
            int r = n - HHALF;
            if (k < HHALF) {
                val = fmaxf(str2m1_w[r * HHALF + k], 0.f);
            } else {
                int c = k - HHALF;
                float w = fmaxf(m12m1_w[r * HHALF + c], 0.f);
                val = (c < NEXC) ? w : -w;                      // @ diag(d)
            }
        }
    } else {
        int ki = (kt - NKT_H) * 32 + klocal;    // 0..63
        val = (n < HHALF) ? inp_weight[ki * HH + n] : 0.f;      // drive mask
    }
    Wpack[idx] = (_Float16)val;
}

// ---------------------------------------------------------------------------
// Kernel 2: f16 conversions: hbuf[0] <- hn, inp16 <- inp
// ---------------------------------------------------------------------------
__global__ void conv_kernel(const float* __restrict__ hn,
                            const float* __restrict__ inp,
                            _Float16* __restrict__ hbuf,
                            _Float16* __restrict__ inp16) {
    long long idx = (long long)blockIdx.x * blockDim.x + threadIdx.x;
    const long long NH = (long long)BB * HH;
    if (idx < NH) {
        hbuf[idx] = (_Float16)hn[idx];
    } else {
        long long j = idx - NH;
        if (j < (long long)BB * TT * II) inp16[j] = (_Float16)inp[j];
    }
}

// ---------------------------------------------------------------------------
// Kernel 3: the 1000-step scan.
// ---------------------------------------------------------------------------
__global__ __launch_bounds__(256, 1)
void rnn_scan_kernel(const _Float16* __restrict__ Wpack,
                     _Float16* __restrict__ hbuf,
                     const _Float16* __restrict__ inp16,
                     const float* __restrict__ x0,
                     float* __restrict__ dout) {
    extern __shared__ __align__(32) _Float16 smem[];
    _Float16* Wlds = smem;                    // [8 local n-tiles][34][512]
    _Float16* Alds = smem + WLDS_ELEMS;       // [34][512]

    const int tid  = threadIdx.x;
    const int wave = tid >> 5;
    const int lane = tid & 31;
    const int wic  = blockIdx.x & 7;          // column slice within cluster
    const int bt   = blockIdx.x >> 3;         // batch tile (cluster)
    const int colbase = wic * 128 + wave * 16;
    // wave-uniform K trip count, forced scalar so the K-loop compiles to a
    // pure SALU-branch loop (ds_load + wmma only) and EXEC stays all-1s by
    // construction at every WMMA.
    const int nkt = __builtin_amdgcn_readfirstlane(
        (colbase < HHALF) ? NKT_ALL : NKT_H);
    const int bRow0 = bt * 16;

    // one-time W slice copy -> LDS (8 contiguous n-tiles of Wpack)
    {
        const uint4* s4 = (const uint4*)(Wpack + (long long)(wic * 8) * NKT_ALL * TILE_E);
        uint4* d4 = (uint4*)Wlds;
        for (int i = tid; i < WLDS_ELEMS / 8; i += 256) d4[i] = s4[i];
    }
    __syncthreads();

    // C/D fragment geometry: lane -> (n, row-base)
    const int nl = lane & 15;
    const int mrow0 = (lane >> 4) << 3;       // 0 or 8
    const int n = colbase + nl;

    float xs[8];
    #pragma unroll
    for (int r = 0; r < 8; ++r)
        xs[r] = x0[(long long)(bRow0 + mrow0 + r) * HH + n];

    float* rnn_out = dout + RNN_OFF;
    float* x_out   = dout + XOUT_OFF;
    const _Float16* wp = Wlds + (long long)wave * NKT_ALL * TILE_E;
    int cur = 0;

    for (int t = 0; t < TT; ++t) {
        // stage A fragments (h tile + inp tile) into LDS in 16 B chunks
        const _Float16* hb = hbuf + (long long)cur * BB * HH;
        for (int c = tid; c < NKT_ALL * 64; c += 256) {
            int kt = c >> 6;
            int j = c & 63;
            int fl = j >> 1, part = j & 1;
            int m = fl & 15, half = fl >> 4;
            int koff = (part ? 16 : 0) + half * 8;
            uint4 v;
            if (kt < NKT_H) {
                v = *(const uint4*)(hb + (long long)(bRow0 + m) * HH + kt * 32 + koff);
            } else {
                const _Float16* ip = inp16 + ((long long)(bRow0 + m) * TT + t) * II
                                     + (kt - NKT_H) * 32 + koff;
                v = *(const uint4*)ip;
                // pull next timestep's input rows toward the WGP (read-once
                // stream; gfx1250 global_prefetch path)
                if (t + 1 < TT) __builtin_prefetch(ip + II, 0, 1);
            }
            *(uint4*)(Alds + kt * TILE_E + fl * 16 + part * 8) = v;
        }
        __syncthreads();

        v8f acc = {};
        #pragma unroll 4
        for (int kt = 0; kt < nkt; ++kt) {
            v16h a = *(const v16h*)(Alds + kt * TILE_E + lane * 16);
            v16h b = *(const v16h*)(wp   + kt * TILE_E + lane * 16);
            acc = __builtin_amdgcn_wmma_f32_16x16x32_f16(
                false, a, false, b, (short)0, acc, false, false);
        }

        // leaky-integrator update + streamed outputs
        _Float16* hb_n = hbuf + (long long)(cur ^ 1) * BB * HH;
        #pragma unroll
        for (int r = 0; r < 8; ++r) {
            int brow = bRow0 + mrow0 + r;
            float xn = 0.9f * xs[r] + 0.1f * acc[r];
            float hnv = xn > 0.f ? xn : 0.f;
            xs[r] = xn;
            long long o = ((long long)brow * TT + t) * HH + n;
            __builtin_nontemporal_store(xn,  &x_out[o]);
            __builtin_nontemporal_store(hnv, &rnn_out[o]);
            hb_n[(long long)brow * HH + n] = (_Float16)hnv;
            if (t == TT - 1) {
                dout[HN_OFF    + (long long)brow * HH + n] = hnv;
                dout[XLAST_OFF + (long long)brow * HH + n] = xn;
            }
        }

        // publish h tile, then cluster-wide step barrier (barrier -3)
        asm volatile("s_wait_storecnt 0" ::: "memory");
        __syncthreads();
        __builtin_amdgcn_s_cluster_barrier();
        cur ^= 1;
    }
}

// ---------------------------------------------------------------------------
// Kernel 4: sigmoid head. One wave per (b,t): dot(h_alm, fc1_w[512:]) + b.
// ---------------------------------------------------------------------------
__global__ void head_kernel(const float* __restrict__ rnn_out,
                            const float* __restrict__ fc1_w,
                            const float* __restrict__ fc1_b,
                            float* __restrict__ out) {
    int gid = blockIdx.x * 8 + (threadIdx.x >> 5);
    if (gid >= BB * TT) return;
    int lane = threadIdx.x & 31;
    int b = gid / TT, t = gid % TT;
    const float* row = rnn_out + ((long long)b * TT + t) * HH + HHALF;
    float s = 0.f;
    #pragma unroll
    for (int j = lane; j < HHALF; j += 32)
        s += __builtin_nontemporal_load(&row[j]) * fc1_w[HHALF + j];
    #pragma unroll
    for (int off = 16; off > 0; off >>= 1) s += __shfl_xor(s, off, 32);
    if (lane == 0) out[gid] = 1.f / (1.f + expf(-(s + fc1_b[0])));
}

// ---------------------------------------------------------------------------
extern "C" void kernel_launch(void* const* d_in, const int* in_sizes, int n_in,
                              void* d_out, int out_size, void* d_ws, size_t ws_size,
                              hipStream_t stream) {
    const float* inp  = (const float*)d_in[0];
    const float* hn0  = (const float*)d_in[1];
    const float* x0   = (const float*)d_in[2];
    const float* s2sf = (const float*)d_in[7];
    const float* s2m  = (const float*)d_in[4];
    const float* m2m  = (const float*)d_in[5];
    const float* m2s  = (const float*)d_in[6];
    const float* wIn  = (const float*)d_in[8];
    const float* fc1w = (const float*)d_in[9];
    const float* fc1b = (const float*)d_in[10];
    float* out = (float*)d_out;

    char* ws = (char*)d_ws;
    _Float16* Wpack = (_Float16*)ws;
    _Float16* hbuf  = (_Float16*)(ws + WPACK_BYTES);
    _Float16* inp16 = (_Float16*)(ws + WPACK_BYTES + HBUF_BYTES);

    pack_w_kernel<<<(int)((WPACK_ELEMS + 255) / 256), 256, 0, stream>>>(
        s2sf, m2s, s2m, m2m, wIn, Wpack);

    long long convN = (long long)BB * HH + (long long)BB * TT * II;
    conv_kernel<<<(int)((convN + 255) / 256), 256, 0, stream>>>(hn0, inp, hbuf, inp16);

    hipFuncSetAttribute((const void*)rnn_scan_kernel,
                        hipFuncAttributeMaxDynamicSharedMemorySize, (int)LDS_BYTES);
    // 8 clusters x 8 WGs (consecutive blockIdx assumed co-clustered; the
    // cluster barrier degrades to NOP if the launch isn't cluster-configured)
    rnn_scan_kernel<<<64, 256, LDS_BYTES, stream>>>(Wpack, hbuf, inp16, x0, out);

    head_kernel<<<(BB * TT) / 8, 256, 0, stream>>>(out + RNN_OFF, fc1w, fc1b, out);
}
